// CCVAE_21715354649885
// MI455X (gfx1250) — matrix operand order repaired and checked by reading, not compiled, not run on hardware
//
#include <hip/hip_runtime.h>
#include <hip/hip_bf16.h>

// ---------------------------------------------------------------------------
// CCVAE forward on gfx1250: bf16 WMMA GEMM chain fed by the Tensor Data Mover
// (TDM), plus softplus/normalize and a rejection sampler.
// ---------------------------------------------------------------------------

typedef __attribute__((ext_vector_type(16))) __bf16 bf16x16;
typedef __attribute__((ext_vector_type(8)))  float  f32x8;
typedef __attribute__((ext_vector_type(4)))  unsigned int u32x4;
typedef __attribute__((ext_vector_type(8)))  int i32x8;
typedef __attribute__((ext_vector_type(4)))  int i32x4;

#define EPSF 1e-8f
#define WAVE 32

#if __has_builtin(__builtin_amdgcn_tensor_load_to_lds)
#define USE_TDM 1
#else
#define USE_TDM 0
#endif

// therock-10.0 headers (6-arg builtin) ship this header; ROCm 7.2 (5-arg) doesn't.
#if __has_include(<hip/amd_detail/amd_gfx1250_TDM.h>)
#define TDM_6ARG 1
#else
#define TDM_6ARG 0
#endif

__device__ __forceinline__ unsigned short f2bf(float f) {
    unsigned int x = __float_as_uint(f);
    unsigned int r = ((x >> 16) & 1u) + 0x7FFFu;   // round-to-nearest-even
    return (unsigned short)((x + r) >> 16);
}

#if USE_TDM
// Issue a TDM 2D tile load: `rows` x 32 bf16 tile from a row-major matrix with
// row stride `stride_elems`, into LDS at byte offset `lds_byte_off` with a
// 16-byte pad after every 64-byte row (LDS row stride = 80 B = 40 bf16).
__device__ __forceinline__ void tdm_load_tile(unsigned lds_byte_off,
                                              const void* gptr,
                                              unsigned rows,
                                              unsigned stride_elems) {
    unsigned long long ga = (unsigned long long)(uintptr_t)gptr;
    const unsigned TD0 = 32u;  // tile K extent (elements)
    // D# group 0: count=1 | lds_addr | global_addr[56:0] | type=2 ("image")
    u32x4 g0 = { 1u,
                 lds_byte_off,
                 (unsigned)ga,
                 (unsigned)((ga >> 32) & 0x01FFFFFFu) | (2u << 30) };
    // D# group 1 bit packing (256b):
    //  [17:16] data_size=1 (2B)   [20] pad_enable
    //  [24:22] pad_interval=3 (16 DWORDs = one 64B row)
    //  [31:25] pad_amount=3 (4 DWORDs = 16B)
    //  [79:48] tensor_dim0 = 32   [111:80] tensor_dim1 = rows
    //  [127:112] tile_dim0 = 32   [143:128] tile_dim1 = rows (tile_dim2 = 0)
    //  [207:160] tensor_dim0_stride = stride_elems
    unsigned w0 = (1u << 16) | (1u << 20) | (3u << 22) | (3u << 25);
    unsigned w1 = (TD0 & 0xFFFFu) << 16;
    unsigned w2 = (TD0 >> 16) | ((rows & 0xFFFFu) << 16);
    unsigned w3 = (rows >> 16) | (TD0 << 16);
    unsigned w4 = rows & 0xFFFFu;
    unsigned w5 = stride_elems;
    i32x8 g1 = { (int)w0, (int)w1, (int)w2, (int)w3,
                 (int)w4, (int)w5, 0, 0 };
    i32x4 z4 = { 0, 0, 0, 0 };
#if TDM_6ARG
    i32x8 z8 = { 0, 0, 0, 0, 0, 0, 0, 0 };
    __builtin_amdgcn_tensor_load_to_lds(g0, g1, z4, z4, z8, 0);
#else
    __builtin_amdgcn_tensor_load_to_lds(g0, g1, z4, z4, 0);
#endif
}

__device__ __forceinline__ unsigned lds_offset_of(const void* p) {
    // generic LDS pointer: low 32 bits are the LDS byte offset
    return (unsigned)(unsigned long long)(uintptr_t)p;
}
#endif  // USE_TDM

// ---------------------------------------------------------------------------
// Prep: fp32 -> bf16 with K padding (zeros in pad region)
// ---------------------------------------------------------------------------
__global__ void convert_pad_bf16(const float* __restrict__ src,
                                 unsigned short* __restrict__ dst,
                                 int M, int K, int Kp) {
    size_t i = (size_t)blockIdx.x * blockDim.x + threadIdx.x;
    size_t total = (size_t)M * Kp;
    if (i >= total) return;
    int r = (int)(i / Kp), c = (int)(i % Kp);
    dst[i] = (c < K) ? f2bf(src[(size_t)r * K + c]) : (unsigned short)0;
}

// Prep: W (K x N fp32, row-major) -> Wt (Np x Kp bf16, row-major), zero-padded
__global__ void transpose_pad_bf16(const float* __restrict__ W,
                                   unsigned short* __restrict__ Wt,
                                   int K, int N, int Kp, int Np) {
    size_t i = (size_t)blockIdx.x * blockDim.x + threadIdx.x;
    size_t total = (size_t)Np * Kp;
    if (i >= total) return;
    int n = (int)(i / Kp), k = (int)(i % Kp);
    Wt[i] = (n < N && k < K) ? f2bf(W[(size_t)k * N + n]) : (unsigned short)0;
}

// ---------------------------------------------------------------------------
// GEMM: out = act(A[M x Kp] @ Wt[Np x Kp]^T + bias)
// A bf16 row-major (lda = Kp), Wt bf16 row-major (ldb = Kp).
// 256 threads (8 waves), block tile 128(M) x 64(N), K-step 32.
// LDS tiles are double-buffered and filled by TDM DMA (wave 0 issues the
// descriptors; completion via TENSORcnt + workgroup barrier). Each wave runs
// 4 x v_wmma_f32_16x16x32_bf16 per K-step (16x64 strip).
// ---------------------------------------------------------------------------
__global__ __launch_bounds__(256)
void gemm_bf16_wmma(const unsigned short* __restrict__ A,
                    const unsigned short* __restrict__ Wt,
                    const float* __restrict__ bias,
                    unsigned short* __restrict__ out_bf16, int ldo_bf,
                    float* __restrict__ out_f32, int ldo_f32,
                    int Nvalid, int Kp, int act) {
    constexpr int BM = 128, BN = 64, BK = 32, LSTR = 40;  // 80B LDS row stride
    __shared__ __align__(16) unsigned short lA[2][BM * LSTR];
    __shared__ __align__(16) unsigned short lB[2][BN * LSTR];

    const int tid  = threadIdx.x;
    const int wave = tid / WAVE;
    const int lane = tid % WAVE;
    const int half = lane >> 4;     // 0: lanes 0-15, 1: lanes 16-31
    const int l15  = lane & 15;
    const int m0   = blockIdx.y * BM;
    const int n0   = blockIdx.x * BN;
    const int nk   = Kp / BK;

    f32x8 acc[4] = {};
    union Frag { bf16x16 v; u32x4 q[2]; };

#if USE_TDM
    const int wid_scalar = __builtin_amdgcn_readfirstlane((int)threadIdx.x) >> 5;
    if (wid_scalar == 0) {
        tdm_load_tile(lds_offset_of(&lA[0][0]), A + (size_t)m0 * Kp, BM, Kp);
        tdm_load_tile(lds_offset_of(&lB[0][0]), Wt + (size_t)n0 * Kp, BN, Kp);
    }
#endif

    for (int kt = 0; kt < nk; ++kt) {
#if USE_TDM
        const int cur = kt & 1;
        __syncthreads();  // prior reads of buf[1-cur] are done before its refill
        if (wid_scalar == 0) {
            if (kt + 1 < nk) {
                int k0n = (kt + 1) * BK;
                tdm_load_tile(lds_offset_of(&lA[1 - cur][0]),
                              A + (size_t)m0 * Kp + k0n, BM, Kp);
                tdm_load_tile(lds_offset_of(&lB[1 - cur][0]),
                              Wt + (size_t)n0 * Kp + k0n, BN, Kp);
                // 2 descriptors for buf[cur] retire in order before these 2
                __builtin_amdgcn_s_wait_tensorcnt((short)2);
            } else {
                __builtin_amdgcn_s_wait_tensorcnt((short)0);
            }
        }
        __syncthreads();  // publish buf[cur] to all waves
#else
        const int cur = 0;
        const int k0 = kt * BK;
        __syncthreads();
        {   // cooperative staging fallback
            int c = tid, r = c >> 2, kc = c & 3;
            *(u32x4*)&lA[0][r * LSTR + kc * 8] =
                *(const u32x4*)&A[(size_t)(m0 + r) * Kp + k0 + kc * 8];
            c = tid + 256; r = c >> 2; kc = c & 3;
            *(u32x4*)&lA[0][r * LSTR + kc * 8] =
                *(const u32x4*)&A[(size_t)(m0 + r) * Kp + k0 + kc * 8];
            int rb = tid >> 2, kb = tid & 3;
            *(u32x4*)&lB[0][rb * LSTR + kb * 8] =
                *(const u32x4*)&Wt[(size_t)(n0 + rb) * Kp + k0 + kb * 8];
        }
        __syncthreads();
#endif

        // A fragment (16x32, ISA layout):
        //  lanes 0-15 : row = l15, elems 0-7 = K 0-7,  elems 8-15 = K 16-23
        //  lanes16-31 : row = l15, elems 0-7 = K 8-15, elems 8-15 = K 24-31
        Frag af;
        {
            const unsigned short* base = &lA[cur][(wave * 16 + l15) * LSTR + half * 8];
            af.q[0] = *(const u32x4*)(base);
            af.q[1] = *(const u32x4*)(base + 16);
        }
        // Preload all 4 B fragments (32x16 each) so ds latency overlaps
        Frag bfr[4];
#pragma unroll
        for (int j = 0; j < 4; ++j) {
            const unsigned short* bbase = &lB[cur][(j * 16 + l15) * LSTR + half * 8];
            bfr[j].q[0] = *(const u32x4*)(bbase);
            bfr[j].q[1] = *(const u32x4*)(bbase + 16);
        }
#pragma unroll
        for (int j = 0; j < 4; ++j) {
            acc[j] = __builtin_amdgcn_wmma_f32_16x16x32_bf16(
                false, af.v, false, bfr[j].v, (short)0, acc[j], false, false);
        }
    }

    // Epilogue: C layout — VGPR i: lanes 0-15 -> M = i, lanes 16-31 -> M = 8+i
    const int mrow = m0 + wave * 16 + half * 8;
#pragma unroll
    for (int j = 0; j < 4; ++j) {
        int col = n0 + j * 16 + l15;
        bool ok = (col < Nvalid);
        float bv = ok ? bias[col] : 0.0f;
#pragma unroll
        for (int i = 0; i < 8; ++i) {
            float v = acc[j][i] + bv;
            if (act == 1) v = fmaxf(v, 0.0f);
            int row = mrow + i;
            if (ok && out_bf16) out_bf16[(size_t)row * ldo_bf + col] = f2bf(v);
            if (ok && out_f32)  out_f32[(size_t)row * ldo_f32 + col] = v;
        }
    }
}

// ---------------------------------------------------------------------------
// softplus -> clamp(EPS) -> row-normalize.  K=64: one wave per row, 2 cols/lane.
// ---------------------------------------------------------------------------
__device__ __forceinline__ float softplus_f(float x) {
    return (x > 20.0f) ? x : log1pf(expf(x));
}

__global__ __launch_bounds__(256)
void lambda_norm_kernel(const float* __restrict__ pre, float* __restrict__ lam) {
    int row  = blockIdx.x * 8 + (threadIdx.x >> 5);
    int lane = threadIdx.x & 31;
    const float* p = pre + (size_t)row * 64;
    float v0 = fmaxf(softplus_f(p[lane]),      EPSF);
    float v1 = fmaxf(softplus_f(p[lane + 32]), EPSF);
    float s = v0 + v1;
#pragma unroll
    for (int m = 16; m > 0; m >>= 1) s += __shfl_xor(s, m, 32);
    float* o = lam + (size_t)row * 64;
    o[lane]      = v0 / s;
    o[lane + 32] = v1 / s;
}

// ---------------------------------------------------------------------------
// Concrete-categorical ordered reparameterized sampler (rejection, 16 attempts)
// One thread per row; hash-based RNG stands in for threefry.
// ---------------------------------------------------------------------------
__device__ __forceinline__ unsigned int hash3(unsigned int a, unsigned int b,
                                              unsigned int c) {
    unsigned int h = a * 0x9E3779B9u ^ b * 0x85EBCA6Bu ^ c * 0xC2B2AE35u;
    h ^= h >> 16; h *= 0x7FEB352Du;
    h ^= h >> 15; h *= 0x846CA68Bu;
    h ^= h >> 16;
    return h;
}

__device__ __forceinline__ float inv_cdf_f(float u, float l) {
    bool nh = (l > 0.499f) && (l < 0.501f);
    float sl = fminf(fmaxf(l, EPSF), 1.0f - EPSF);
    if (nh) sl = 0.4f;
    u = fminf(fmaxf(u, EPSF), 1.0f - EPSF);
    float num = logf(u * (2.0f * sl - 1.0f) + 1.0f - sl) - logf(1.0f - sl);
    float den = logf(sl) - logf(1.0f - sl);
    return nh ? u : num / den;
}

__global__ __launch_bounds__(256)
void sampler_kernel(const float* __restrict__ lam, float* __restrict__ z_f32,
                    unsigned short* __restrict__ z_bf16, int M) {
    int row = blockIdx.x * blockDim.x + threadIdx.x;
    if (row >= M) return;

    float v[64]; int ord[64];
#pragma unroll 4
    for (int i = 0; i < 64; ++i) { v[i] = lam[(size_t)row * 64 + i]; ord[i] = i; }

    // argsort descending (selection sort)
    for (int i = 0; i < 63; ++i) {
        int mi = i;
        for (int j = i + 1; j < 64; ++j) if (v[j] > v[mi]) mi = j;
        float tv = v[i]; v[i] = v[mi]; v[mi] = tv;
        int ti = ord[i]; ord[i] = ord[mi]; ord[mi] = ti;
    }

    float lam1 = v[0];
    float cb[63], fin[63];
    for (int j = 0; j < 63; ++j) {
        cb[j]  = v[j + 1] / (v[j + 1] + lam1 + EPSF);
        fin[j] = 0.0f;
    }

    bool active = true;
    for (int a = 0; a < 16; ++a) {
        float cand[63], s = 0.0f;
        for (int j = 0; j < 63; ++j) {
            float u = (float)(hash3((unsigned)row, (unsigned)a, (unsigned)j) >> 8) *
                      (1.0f / 16777216.0f);
            cand[j] = inv_cdf_f(u, cb[j]);
            s += cand[j];
        }
        bool acc = active && (s <= 1.0f);
        if (acc) { for (int j = 0; j < 63; ++j) fin[j] = cand[j]; }
        active = active && !acc;
    }

    float s = 0.0f;
    for (int j = 0; j < 63; ++j) s += fin[j];
    float x1 = fmaxf(1.0f - s, EPSF);

    // inverse permutation: out[order[i]] = x_sorted[i]
    float* zo = z_f32 + (size_t)row * 64;
    unsigned short* zb = z_bf16 + (size_t)row * 64;
    zo[ord[0]] = x1;            zb[ord[0]] = f2bf(x1);
    for (int j = 0; j < 63; ++j) {
        zo[ord[j + 1]] = fin[j]; zb[ord[j + 1]] = f2bf(fin[j]);
    }
}

// ---------------------------------------------------------------------------
// Orchestration
// ---------------------------------------------------------------------------
extern "C" void kernel_launch(void* const* d_in, const int* in_sizes, int n_in,
                              void* d_out, int out_size, void* d_ws, size_t ws_size,
                              hipStream_t stream) {
    (void)in_sizes; (void)n_in; (void)out_size; (void)ws_size;
    const int M = 32768, D = 784, H1 = 512, H2 = 256, K = 64;
    const int Dp = 800;   // 784 padded to mult of 32
    const int Np3 = 832;  // 784 padded to mult of 64 (decoder output tiles)

    const float* x      = (const float*)d_in[0];
    const float* eW1    = (const float*)d_in[1];
    const float* eb1    = (const float*)d_in[2];
    const float* eW2    = (const float*)d_in[3];
    const float* eb2    = (const float*)d_in[4];
    const float* lW     = (const float*)d_in[5];
    const float* lb     = (const float*)d_in[6];
    const float* dW1    = (const float*)d_in[7];
    const float* db1    = (const float*)d_in[8];
    const float* dW2    = (const float*)d_in[9];
    const float* db2    = (const float*)d_in[10];
    const float* dW3    = (const float*)d_in[11];
    const float* db3    = (const float*)d_in[12];

    float* out = (float*)d_out;
    float* out_logits = out;                                   // M*784
    float* out_z      = out + (size_t)M * D;                   // M*64
    float* out_lam    = out + (size_t)M * D + (size_t)M * K;   // M*64

    // workspace carve-up (256B aligned)
    char* ws = (char*)d_ws;
    size_t off = 0;
    auto carve = [&](size_t bytes) -> char* {
        char* p = ws + off;
        off += (bytes + 255) & ~(size_t)255;
        return p;
    };
    unsigned short* x_bf   = (unsigned short*)carve((size_t)M * Dp * 2);
    unsigned short* W1T    = (unsigned short*)carve((size_t)H1 * Dp * 2);
    unsigned short* W2T    = (unsigned short*)carve((size_t)H2 * H1 * 2);
    unsigned short* LWT    = (unsigned short*)carve((size_t)K * H2 * 2);
    unsigned short* D1T    = (unsigned short*)carve((size_t)H2 * K * 2);
    unsigned short* D2T    = (unsigned short*)carve((size_t)H1 * H2 * 2);
    unsigned short* D3T    = (unsigned short*)carve((size_t)Np3 * H1 * 2);
    unsigned short* h1     = (unsigned short*)carve((size_t)M * H1 * 2);
    unsigned short* h2     = (unsigned short*)carve((size_t)M * H2 * 2);
    float*          lampre = (float*)        carve((size_t)M * K * 4);
    unsigned short* z_bf   = (unsigned short*)carve((size_t)M * K * 2);
    unsigned short* d1     = (unsigned short*)carve((size_t)M * H2 * 2);
    unsigned short* d2     = (unsigned short*)carve((size_t)M * H1 * 2);

    auto cdiv = [](size_t a, size_t b) { return (unsigned)((a + b - 1) / b); };

    // --- prep: conversions & weight transposes ---
    convert_pad_bf16<<<cdiv((size_t)M * Dp, 256), 256, 0, stream>>>(x, x_bf, M, D, Dp);
    transpose_pad_bf16<<<cdiv((size_t)H1 * Dp, 256), 256, 0, stream>>>(eW1, W1T, D, H1, Dp, H1);
    transpose_pad_bf16<<<cdiv((size_t)H2 * H1, 256), 256, 0, stream>>>(eW2, W2T, H1, H2, H1, H2);
    transpose_pad_bf16<<<cdiv((size_t)K * H2, 256), 256, 0, stream>>>(lW, LWT, H2, K, H2, K);
    transpose_pad_bf16<<<cdiv((size_t)H2 * K, 256), 256, 0, stream>>>(dW1, D1T, K, H2, K, H2);
    transpose_pad_bf16<<<cdiv((size_t)H1 * H2, 256), 256, 0, stream>>>(dW2, D2T, H2, H1, H2, H1);
    transpose_pad_bf16<<<cdiv((size_t)Np3 * H1, 256), 256, 0, stream>>>(dW3, D3T, H1, D, H1, Np3);

    // --- encoder ---
    gemm_bf16_wmma<<<dim3(H1 / 64, M / 128), 256, 0, stream>>>(
        x_bf, W1T, eb1, h1, H1, nullptr, 0, H1, Dp, 1);
    gemm_bf16_wmma<<<dim3(H2 / 64, M / 128), 256, 0, stream>>>(
        h1, W2T, eb2, h2, H2, nullptr, 0, H2, H1, 1);
    gemm_bf16_wmma<<<dim3(K / 64, M / 128), 256, 0, stream>>>(
        h2, LWT, lb, nullptr, 0, lampre, K, K, H2, 0);

    // --- lambda normalize + sample ---
    lambda_norm_kernel<<<M / 8, 256, 0, stream>>>(lampre, out_lam);
    sampler_kernel<<<M / 256, 256, 0, stream>>>(out_lam, out_z, z_bf, M);

    // --- decoder ---
    gemm_bf16_wmma<<<dim3(H2 / 64, M / 128), 256, 0, stream>>>(
        z_bf, D1T, db1, d1, H2, nullptr, 0, H2, K, 1);
    gemm_bf16_wmma<<<dim3(H1 / 64, M / 128), 256, 0, stream>>>(
        d1, D2T, db2, d2, H1, nullptr, 0, H1, H2, 1);
    gemm_bf16_wmma<<<dim3(Np3 / 64, M / 128), 256, 0, stream>>>(
        d2, D3T, db3, nullptr, 0, out_logits, D, D, H1, 0);
}
